// MultiHeadAttention_66580583023121
// MI455X (gfx1250) — matrix operand order repaired and checked by reading, not compiled, not run on hardware
//
#include <hip/hip_runtime.h>
#include <hip/hip_bf16.h>

// ---------------------------------------------------------------------------
// MHA forward for MI455X (gfx1250): bf16 WMMA, f32 accumulate, double-buffered
// LDS staging. B=2, S=2048, D=1024, H=16, HD=64
// ---------------------------------------------------------------------------

typedef __attribute__((ext_vector_type(16))) __bf16 v16bf;
typedef __attribute__((ext_vector_type(8)))  float  v8f;

#define BATCH 2
#define SEQ   2048
#define DIM   1024
#define NH    16
#define HD    64

static __device__ inline unsigned short f2bf(float f) {
    union { float f; unsigned int u; } x; x.f = f;
    unsigned int u = x.u;
    unsigned int r = (u + 0x7FFFu + ((u >> 16) & 1u)) >> 16; // RNE
    return (unsigned short)r;
}

static __device__ inline __bf16 asbf(unsigned short u) {
    union { unsigned short u; __bf16 b; } x; x.u = u;
    return x.b;
}

// ---------------------------------------------------------------------------
// f32 -> bf16 conversion (grid-stride)
// ---------------------------------------------------------------------------
__global__ __launch_bounds__(256) void cvt_bf16_kernel(
    const float* __restrict__ src, unsigned short* __restrict__ dst, int n)
{
    int i = blockIdx.x * blockDim.x + threadIdx.x;
    int stride = gridDim.x * blockDim.x;
    for (; i < n; i += stride) dst[i] = f2bf(src[i]);
}

// ---------------------------------------------------------------------------
// GEMM: C[M,N] = A[M,K](bf16 rm) * W[N,K]^T(bf16) + bias[N]
// mode 0: f32 -> outF[M,N];  mode 1: bf16 scatter to qkv [3][B,H,S,HD]
// Block tile 128(M) x 128(N), 8 waves as 2(M) x 4(N); wave tile 64x32
// (4x2 WMMA tiles = 8 WMMAs per 32-deep K step). Double-buffered LDS.
// ---------------------------------------------------------------------------
__global__ __launch_bounds__(256) void gemm_bf16_kernel(
    const unsigned short* __restrict__ A,
    const unsigned short* __restrict__ W,
    const float* __restrict__ bias,
    float* __restrict__ outF,
    unsigned short* __restrict__ qkvOut,
    int M, int N, int K, int mode)
{
    __shared__ unsigned short Al[2][128 * 32];
    __shared__ unsigned short Bl[2][128 * 32];

    const int tid   = threadIdx.x;
    const int wave  = tid >> 5;
    const int lane  = tid & 31;
    const int half  = lane >> 4;
    const int l15   = lane & 15;
    const int waveM = wave >> 2;   // 0..1 -> 64 rows each
    const int waveN = wave & 3;    // 0..3 -> 32 cols each
    const int blockM = blockIdx.y * 128;
    const int blockN = blockIdx.x * 128;
    const int Ku = K >> 1;         // row length in uints

    const int kb0 = half * 8;
    const int kb1 = 16 + half * 8;

    // cooperative-load geometry: 128x32 bf16 = 2048 uints, 8 per thread,
    // each thread reads 8 consecutive uints (32B) of one row.
    const int rL = tid >> 1;
    const int cL = (tid & 1) * 8;
    const unsigned int* srcA = (const unsigned int*)A;
    const unsigned int* srcB = (const unsigned int*)W;

    v8f acc[4][2];
#pragma unroll
    for (int mi = 0; mi < 4; ++mi)
#pragma unroll
        for (int ni = 0; ni < 2; ++ni) acc[mi][ni] = (v8f){};

    unsigned int pa[8], pb[8];
    const int kTiles = K >> 5;

    // prologue: tile 0 -> buf 0
    {
        size_t baseA = (size_t)(blockM + rL) * Ku + cL;
        size_t baseB = (size_t)(blockN + rL) * Ku + cL;
#pragma unroll
        for (int i = 0; i < 8; ++i) { pa[i] = srcA[baseA + i]; pb[i] = srcB[baseB + i]; }
        unsigned int* dA = (unsigned int*)&Al[0][0];
        unsigned int* dB = (unsigned int*)&Bl[0][0];
#pragma unroll
        for (int i = 0; i < 8; ++i) { dA[tid * 8 + i] = pa[i]; dB[tid * 8 + i] = pb[i]; }
    }
    __syncthreads();

    for (int kt = 0; kt < kTiles; ++kt) {
        const int cur = kt & 1;
        const bool more = (kt + 1) < kTiles;
        if (more) {
            size_t baseA = (size_t)(blockM + rL) * Ku + (kt + 1) * 16 + cL;
            size_t baseB = (size_t)(blockN + rL) * Ku + (kt + 1) * 16 + cL;
#pragma unroll
            for (int i = 0; i < 8; ++i) { pa[i] = srcA[baseA + i]; pb[i] = srcB[baseB + i]; }
        }

        v16bf afr[4], bfr[2];
#pragma unroll
        for (int mi = 0; mi < 4; ++mi) {
            const unsigned short* Ar = &Al[cur][(waveM * 64 + mi * 16 + l15) * 32];
#pragma unroll
            for (int i = 0; i < 8; ++i) {
                afr[mi][i]     = asbf(Ar[half * 16 + i]);
                afr[mi][8 + i] = asbf(Ar[half * 16 + 8 + i]);
            }
        }
#pragma unroll
        for (int ni = 0; ni < 2; ++ni) {
            const unsigned short* Br = &Bl[cur][(waveN * 32 + ni * 16 + l15) * 32];
#pragma unroll
            for (int v = 0; v < 8; ++v) {
                bfr[ni][2 * v]     = asbf(Br[half * 16 + 2 * v]);
                bfr[ni][2 * v + 1] = asbf(Br[half * 16 + 2 * v + 1]);
            }
        }
#pragma unroll
        for (int mi = 0; mi < 4; ++mi)
#pragma unroll
            for (int ni = 0; ni < 2; ++ni)
                acc[mi][ni] = __builtin_amdgcn_wmma_f32_16x16x32_bf16(
                    false, afr[mi], false, bfr[ni], (short)0, acc[mi][ni], false, false);

        if (more) {
            unsigned int* dA = (unsigned int*)&Al[cur ^ 1][0];
            unsigned int* dB = (unsigned int*)&Bl[cur ^ 1][0];
#pragma unroll
            for (int i = 0; i < 8; ++i) { dA[tid * 8 + i] = pa[i]; dB[tid * 8 + i] = pb[i]; }
        }
        __syncthreads();
    }

    // epilogue
#pragma unroll
    for (int mi = 0; mi < 4; ++mi) {
#pragma unroll
        for (int ni = 0; ni < 2; ++ni) {
#pragma unroll
            for (int r = 0; r < 8; ++r) {
                int row = blockM + waveM * 64 + mi * 16 + half * 8 + r;
                int col = blockN + waveN * 32 + ni * 16 + l15;
                float v = acc[mi][ni][r] + bias[col];
                if (mode == 0) {
                    outF[(size_t)row * N + col] = v;
                } else {
                    int which = col >> 10;      // 0:Q 1:K 2:V
                    int dfeat = col & 1023;
                    int h  = dfeat >> 6;
                    int hd = dfeat & 63;
                    int bidx = row >> 11;       // / SEQ
                    int s    = row & 2047;
                    size_t off = (size_t)which * ((size_t)BATCH * NH * SEQ * HD)
                               + (((size_t)(bidx * NH + h) * SEQ + s) * HD + hd);
                    qkvOut[off] = f2bf(v);
                }
            }
        }
    }
}

// ---------------------------------------------------------------------------
// Causal flash attention. Grid: (SEQ/256, B*NH). 8 waves; wave owns 32 q-rows
// (two 16-row tiles). Per 32-key step: 8 score WMMAs + 8 PV WMMAs; K/V
// fragment builds shared across the two q tiles. Double-buffered K/V in LDS.
// ---------------------------------------------------------------------------
__global__ __launch_bounds__(256) void attn_kernel(
    const unsigned short* __restrict__ Qw,   // [B*H][S][HD]
    const unsigned short* __restrict__ Kw,
    const unsigned short* __restrict__ Vw,
    unsigned short* __restrict__ Ow)         // [B*S][D], head-concat columns
{
    __shared__ unsigned short Kl[2][32 * 64];
    __shared__ unsigned short Vl[2][32 * 64];
    __shared__ unsigned short Pl[8][2][16 * 32];

    const int tid  = threadIdx.x;
    const int wave = tid >> 5;
    const int lane = tid & 31;
    const int half = lane >> 4;
    const int l15  = lane & 15;
    const int bh   = blockIdx.y;             // b*NH + h
    const int b    = bh >> 4;
    const int h    = bh & 15;
    const int qBase = blockIdx.x * 256;
    const int qRow0 = qBase + wave * 32;

    const unsigned short* Qbh = Qw + (size_t)bh * (SEQ * HD);
    const unsigned short* Kbh = Kw + (size_t)bh * (SEQ * HD);
    const unsigned short* Vbh = Vw + (size_t)bh * (SEQ * HD);

    const int kb0 = half * 8;
    const int kb1 = 16 + half * 8;

    // Q fragments: two 16-row tiles, each split into two 16x32 A-fragments
    v16bf qf[2][2];
#pragma unroll
    for (int qt = 0; qt < 2; ++qt) {
        const unsigned short* Qr = Qbh + (size_t)(qRow0 + qt * 16 + l15) * HD;
#pragma unroll
        for (int f = 0; f < 2; ++f)
#pragma unroll
            for (int i = 0; i < 8; ++i) {
                qf[qt][f][i]     = asbf(Qr[f * 32 + half * 16 + i]);
                qf[qt][f][8 + i] = asbf(Qr[f * 32 + half * 16 + 8 + i]);
            }
    }

    v8f o[2][4];
#pragma unroll
    for (int qt = 0; qt < 2; ++qt)
#pragma unroll
        for (int t = 0; t < 4; ++t) o[qt][t] = (v8f){};
    float mrow[2][8], lrow[2][8];
#pragma unroll
    for (int qt = 0; qt < 2; ++qt)
#pragma unroll
        for (int r = 0; r < 8; ++r) { mrow[qt][r] = -1e30f; lrow[qt][r] = 0.0f; }

    const float scale = 0.125f;  // 1/sqrt(64)
    const int jEnd = qBase + 256;

    unsigned int ka[4], va[4];

    // prologue: K/V block 0 -> buf 0 (32x64 bf16 = 1024 uints each, 4/thread)
    {
        const unsigned int* ks = (const unsigned int*)Kbh;
        const unsigned int* vs = (const unsigned int*)Vbh;
        unsigned int* kd = (unsigned int*)&Kl[0][0];
        unsigned int* vd = (unsigned int*)&Vl[0][0];
#pragma unroll
        for (int i = 0; i < 4; ++i) {
            int idx = tid * 4 + i;
            kd[idx] = ks[idx];
            vd[idx] = vs[idx];
        }
    }
    __syncthreads();

    for (int jb = 0; jb < jEnd; jb += 32) {
        const int cur = (jb >> 5) & 1;
        const bool more = (jb + 32) < jEnd;
        if (more) {
            const unsigned int* ks = (const unsigned int*)(Kbh + (size_t)(jb + 32) * HD);
            const unsigned int* vs = (const unsigned int*)(Vbh + (size_t)(jb + 32) * HD);
#pragma unroll
            for (int i = 0; i < 4; ++i) {
                int idx = tid * 4 + i;
                ka[i] = ks[idx];
                va[i] = vs[idx];
            }
        }

        // scores: per q tile, S[16q x 32k]; K fragments shared across q tiles
        v8f sAcc[2][2];
        sAcc[0][0] = (v8f){}; sAcc[0][1] = (v8f){};
        sAcc[1][0] = (v8f){}; sAcc[1][1] = (v8f){};
#pragma unroll
        for (int t = 0; t < 2; ++t) {
#pragma unroll
            for (int hc = 0; hc < 2; ++hc) {
                v16bf bf;
                const unsigned short* Krow = &Kl[cur][(t * 16 + l15) * 64 + hc * 32];
#pragma unroll
                for (int v = 0; v < 8; ++v) {
                    bf[2 * v]     = asbf(Krow[half * 16 + 2 * v]);
                    bf[2 * v + 1] = asbf(Krow[half * 16 + 2 * v + 1]);
                }
#pragma unroll
                for (int qt = 0; qt < 2; ++qt)
                    sAcc[qt][t] = __builtin_amdgcn_wmma_f32_16x16x32_bf16(
                        false, qf[qt][hc], false, bf, (short)0, sAcc[qt][t], false, false);
            }
        }

        // online softmax update per q tile
#pragma unroll
        for (int qt = 0; qt < 2; ++qt) {
#pragma unroll
            for (int r = 0; r < 8; ++r) {
                int qrow = qRow0 + qt * 16 + half * 8 + r;
                float s0 = sAcc[qt][0][r] * scale;
                float s1 = sAcc[qt][1][r] * scale;
                if (jb + l15 > qrow)      s0 = -1e30f;
                if (jb + 16 + l15 > qrow) s1 = -1e30f;
                float mloc = fmaxf(s0, s1);
#pragma unroll
                for (int off = 1; off < 16; off <<= 1)
                    mloc = fmaxf(mloc, __shfl_xor(mloc, off, 32));
                float mnew  = fmaxf(mrow[qt][r], mloc);
                float alpha = __expf(mrow[qt][r] - mnew);
                float p0 = __expf(s0 - mnew);
                float p1 = __expf(s1 - mnew);
                float ls = p0 + p1;
#pragma unroll
                for (int off = 1; off < 16; off <<= 1)
                    ls += __shfl_xor(ls, off, 32);
                lrow[qt][r] = lrow[qt][r] * alpha + ls;
                mrow[qt][r] = mnew;
#pragma unroll
                for (int t = 0; t < 4; ++t) o[qt][t][r] *= alpha;
                Pl[wave][qt][(half * 8 + r) * 32 + l15]      = f2bf(p0);
                Pl[wave][qt][(half * 8 + r) * 32 + 16 + l15] = f2bf(p1);
            }
        }

        // O += P(16x32) * V(32x64); V fragments shared across q tiles
        v16bf pf[2];
#pragma unroll
        for (int qt = 0; qt < 2; ++qt) {
            const unsigned short* Pr = &Pl[wave][qt][l15 * 32];
#pragma unroll
            for (int i = 0; i < 8; ++i) {
                pf[qt][i]     = asbf(Pr[half * 16 + i]);
                pf[qt][8 + i] = asbf(Pr[half * 16 + 8 + i]);
            }
        }
#pragma unroll
        for (int ct = 0; ct < 4; ++ct) {
            v16bf vf;
#pragma unroll
            for (int v = 0; v < 8; ++v) {
                const unsigned short* Vr = &Vl[cur][(half * 16 + 2 * v) * 64 + ct * 16 + l15];
                vf[2 * v]     = asbf(Vr[0]);
                vf[2 * v + 1] = asbf(Vr[64]);
            }
#pragma unroll
            for (int qt = 0; qt < 2; ++qt)
                o[qt][ct] = __builtin_amdgcn_wmma_f32_16x16x32_bf16(
                    false, pf[qt], false, vf, (short)0, o[qt][ct], false, false);
        }

        if (more) {
            unsigned int* kd = (unsigned int*)&Kl[cur ^ 1][0];
            unsigned int* vd = (unsigned int*)&Vl[cur ^ 1][0];
#pragma unroll
            for (int i = 0; i < 4; ++i) {
                int idx = tid * 4 + i;
                kd[idx] = ka[i];
                vd[idx] = va[i];
            }
        }
        __syncthreads();
    }

    // normalize and store attn output as [B*S, D] bf16 (head-concat columns)
#pragma unroll
    for (int qt = 0; qt < 2; ++qt) {
#pragma unroll
        for (int r = 0; r < 8; ++r) {
            float inv = 1.0f / lrow[qt][r];
            int qrow = qRow0 + qt * 16 + half * 8 + r;
            size_t base = ((size_t)(b * SEQ + qrow)) * DIM + h * HD;
#pragma unroll
            for (int ct = 0; ct < 4; ++ct)
                Ow[base + ct * 16 + l15] = f2bf(o[qt][ct][r] * inv);
        }
    }
}

// ---------------------------------------------------------------------------
// launch
// ---------------------------------------------------------------------------
extern "C" void kernel_launch(void* const* d_in, const int* in_sizes, int n_in,
                              void* d_out, int out_size, void* d_ws, size_t ws_size,
                              hipStream_t stream)
{
    const float* X     = (const float*)d_in[0];   // [B,S,D]
    const float* Wqkv  = (const float*)d_in[1];   // [3D,D]
    const float* bqkv  = (const float*)d_in[2];   // [3D]
    const float* Wout  = (const float*)d_in[3];   // [D,D]
    const float* bout  = (const float*)d_in[4];   // [D]
    float* out = (float*)d_out;                   // [B,S,D] f32

    const size_t MS   = (size_t)BATCH * SEQ;           // 4096
    const size_t qkvC = (size_t)BATCH * NH * SEQ * HD; // per-matrix qkv chunk

    char* p = (char*)d_ws;
    unsigned short* Xb   = (unsigned short*)p; p += MS * DIM * 2;
    unsigned short* Wqb  = (unsigned short*)p; p += (size_t)3 * DIM * DIM * 2;
    unsigned short* Wob  = (unsigned short*)p; p += (size_t)DIM * DIM * 2;
    unsigned short* QKVb = (unsigned short*)p; p += 3 * qkvC * 2;
    unsigned short* Ab   = (unsigned short*)p; p += MS * DIM * 2;

    unsigned short* Qb = QKVb;
    unsigned short* Kb = QKVb + qkvC;
    unsigned short* Vb = QKVb + 2 * qkvC;

    // 1) fp32 -> bf16
    cvt_bf16_kernel<<<2048, 256, 0, stream>>>(X, Xb, (int)(MS * DIM));
    cvt_bf16_kernel<<<2048, 256, 0, stream>>>(Wqkv, Wqb, 3 * DIM * DIM);
    cvt_bf16_kernel<<<1024, 256, 0, stream>>>(Wout, Wob, DIM * DIM);

    // 2) QKV projection: [4096,1024] x [3072,1024]^T -> scatter bf16 q/k/v
    gemm_bf16_kernel<<<dim3(3 * DIM / 128, MS / 128), 256, 0, stream>>>(
        Xb, Wqb, bqkv, nullptr, QKVb, (int)MS, 3 * DIM, DIM, 1);

    // 3) causal flash attention
    attn_kernel<<<dim3(SEQ / 256, BATCH * NH), 256, 0, stream>>>(Qb, Kb, Vb, Ab);

    // 4) output projection: [4096,1024] x [1024,1024]^T + bias -> f32 out
    gemm_bf16_kernel<<<dim3(DIM / 128, MS / 128), 256, 0, stream>>>(
        Ab, Wob, bout, out, nullptr, (int)MS, DIM, DIM, 0);
}
